// ModConv2d_36850819400574
// MI455X (gfx1250) — matrix operand order repaired and checked by reading, not compiled, not run on hardware
//
#include <hip/hip_runtime.h>
#include <hip/hip_bf16.h>
#include <math.h>

// ---------------- types for WMMA ----------------
typedef __attribute__((ext_vector_type(16))) __bf16        v16bf;
typedef __attribute__((ext_vector_type(8)))  float         v8f;
typedef __attribute__((ext_vector_type(8)))  unsigned short u16x8;

union ABFrag {
    u16x8 u[2];
    v16bf v;
};

__device__ __forceinline__ unsigned short f2bf(float f) {
    union { float f; unsigned u; } c; c.f = f;
    unsigned u = c.u;
    unsigned r = (u + 0x7FFFu + ((u >> 16) & 1u)) >> 16;
    return (unsigned short)r;
}

// ---------------- problem dims ----------------
#define NB   16
#define CIN  128
#define COUT 128
#define HH   128
#define WW   128
#define AUX  64
#define HID  256

// ---------------- kernel 1: h = prelu(y @ fc1_w^T + fc1_b) ----------------
__global__ __launch_bounds__(256)
void mlp1_kernel(const float* __restrict__ y, const float* __restrict__ fc1w,
                 const float* __restrict__ fc1b, const float* __restrict__ prelu1,
                 float* __restrict__ hbuf) {
    int n = blockIdx.x;        // 0..15
    int j = threadIdx.x;       // 0..255
    const float* yr = y + n * AUX;
    const float* wr = fc1w + j * AUX;
    float acc = fc1b[j];
#pragma unroll 8
    for (int k = 0; k < AUX; ++k) acc += yr[k] * wr[k];
    float a = prelu1[0];
    hbuf[n * HID + j] = (acc >= 0.f) ? acc : a * acc;
}

// ---------------- kernel 2: m = sigmoid(prelu(h @ fc2_w^T + fc2_b));
//                  wmod[(n,r,s)][co][ci] = bf16(m * weight[co][ci][r][s]) ----
__global__ __launch_bounds__(256)
void mlp2_kernel(const float* __restrict__ hbuf, const float* __restrict__ fc2w,
                 const float* __restrict__ fc2b, const float* __restrict__ prelu2,
                 const float* __restrict__ weight, unsigned short* __restrict__ wmod) {
    int t    = blockIdx.x * 256 + threadIdx.x;  // 0 .. 16*128*128-1
    int nn   = t & 15;                          // sample (fast => fc2w row shared in wave)
    int pair = t >> 4;                          // co*128 + ci
    const float* hr = hbuf + nn * HID;
    const float* wr = fc2w + (size_t)pair * HID;
    float acc = fc2b[pair];
#pragma unroll 8
    for (int k = 0; k < HID; ++k) acc += hr[k] * wr[k];
    float a = prelu2[0];
    acc = (acc >= 0.f) ? acc : a * acc;
    float m = 1.f / (1.f + expf(-acc));
#pragma unroll
    for (int r = 0; r < 3; ++r) {
#pragma unroll
        for (int s = 0; s < 3; ++s) {
            float wv = m * weight[((size_t)pair * 3 + r) * 3 + s];
            wmod[(size_t)((nn * 3 + r) * 3 + s) * (COUT * CIN) + pair] = f2bf(wv);
        }
    }
}

// ---------------- kernel 3: implicit-GEMM conv via bf16 WMMA ----------------
// One workgroup per (n, h): computes out[n][0..127][h][0..127].
// 8 waves mapped 4 (co blocks of 32) x 2 (pixel halves of 64).
// Each wave: 2 co-tiles x 4 pixel-groups = 8 C tiles; each B fragment feeds
// 2 WMMAs (both co-tiles) -> 1 ds_load_b128 per WMMA instead of 2.
#define PITCH 40   // LDS row pitch in ushort (32 ci + pad), 80B rows -> 16B aligned sub-chunks

__global__ __launch_bounds__(256)
void conv_wmma_kernel(const float* __restrict__ x, const unsigned short* __restrict__ wmod,
                      const float* __restrict__ bias, float* __restrict__ out) {
    __shared__ __align__(16) unsigned short xt[130 * PITCH]; // [w+1 : 0..129][ci : 0..31]

    const int bid    = blockIdx.x;      // 0..2047
    const int n      = bid >> 7;        // sample
    const int h      = bid & 127;       // output row
    const int tid    = threadIdx.x;
    const int lane   = tid & 31;
    const int wave   = tid >> 5;        // 0..7
    const int wco    = wave & 3;        // co block (32 rows)
    const int wpx    = wave >> 2;       // pixel half (64 px)
    const int cobase = wco * 32;
    const int pxbase = wpx * 64;
    const int lrow   = lane & 15;       // row/column-in-tile selector
    const int lhi    = lane >> 4;       // 0 or 1

    // zero halo columns (w = -1 and w = 128) once; visible after first fill barrier
    if (tid < 64) {
        int rr = tid >> 5;              // 0 -> w=-1 row, 1 -> w=128 row
        int ci = tid & 31;
        xt[(rr ? 129 : 0) * PITCH + ci] = 0;
    }

    v8f acc[8];                         // [t*4 + p]: t = co tile (0,1), p = px group (0..3)
#pragma unroll
    for (int i = 0; i < 8; ++i) {
        v8f z = {0.f, 0.f, 0.f, 0.f, 0.f, 0.f, 0.f, 0.f};
        acc[i] = z;
    }

    const float* xn = x + (size_t)n * CIN * HH * WW;

    for (int r = 0; r < 3; ++r) {
        const int hh   = h + r - 1;
        const bool inb = (hh >= 0) && (hh < HH);
        for (int c = 0; c < 4; ++c) {           // ci chunks of 32
            __syncthreads();                    // WAR: previous reads done before refill
            // ---- fill LDS transposed: xt[w+1][ci_local], bf16 ----
            {
                const int w   = tid & 127;
                const int cig = tid >> 7;       // 0 or 1 -> ci_local block of 16
                __align__(16) unsigned short vals[16];
                if (inb) {
                    const float* src = xn + ((size_t)(c * 32 + cig * 16) * HH + hh) * WW + w;
#pragma unroll
                    for (int j = 0; j < 16; ++j)
                        vals[j] = f2bf(src[(size_t)j * HH * WW]);
                } else {
#pragma unroll
                    for (int j = 0; j < 16; ++j) vals[j] = 0;
                }
                u16x8* dst = (u16x8*)&xt[(w + 1) * PITCH + cig * 16];
                dst[0] = *(const u16x8*)&vals[0];
                dst[1] = *(const u16x8*)&vals[8];
            }
            __syncthreads();                    // RAW: fill visible before WMMA reads

#pragma unroll
            for (int s = 0; s < 3; ++s) {
                // ---- A fragments for both co tiles ----
                // 16-bit A layout: lanes 0-15 hold K {0-7,16-23}, lanes 16-31 hold K {8-15,24-31}
                ABFrag a[2];
#pragma unroll
                for (int t = 0; t < 2; ++t) {
                    const unsigned short* ap = wmod
                        + (size_t)((n * 3 + r) * 3 + s) * (COUT * CIN)
                        + (size_t)(cobase + t * 16 + lrow) * CIN
                        + c * 32 + lhi * 8;
                    a[t].u[0] = *(const u16x8*)(ap);
                    a[t].u[1] = *(const u16x8*)(ap + 16);
                }

#pragma unroll
                for (int p = 0; p < 4; ++p) {
                    // ---- B fragment: column w = pxbase + p*16 + lrow + (s-1)
                    // 16-bit B layout: lanes 0-15 hold K {0-7,16-23}, lanes 16-31 hold K {8-15,24-31}
                    const int w = pxbase + p * 16 + lrow + s - 1;  // halo zeroed
                    const unsigned short* bp = &xt[(w + 1) * PITCH + lhi * 8];
                    ABFrag b;
                    b.u[0] = *(const u16x8*)(bp);
                    b.u[1] = *(const u16x8*)(bp + 16);

                    acc[0 * 4 + p] = __builtin_amdgcn_wmma_f32_16x16x32_bf16(
                        false, a[0].v, false, b.v, (short)0, acc[0 * 4 + p], false, false);
                    acc[1 * 4 + p] = __builtin_amdgcn_wmma_f32_16x16x32_bf16(
                        false, a[1].v, false, b.v, (short)0, acc[1 * 4 + p], false, false);
                }
            }
        }
    }

    // ---- epilogue: add bias, store f32 ----
    // C layout: VGPR j -> M = j (lanes 0-15) / j+8 (lanes 16-31); N = lane&15
#pragma unroll
    for (int t = 0; t < 2; ++t) {
#pragma unroll
        for (int j = 0; j < 8; ++j) {
            const int co = cobase + t * 16 + j + lhi * 8;
            const float bv = bias[co];
            float* orow = out + (((size_t)(n * COUT + co)) * HH + h) * WW;
#pragma unroll
            for (int p = 0; p < 4; ++p) {
                orow[pxbase + p * 16 + lrow] = acc[t * 4 + p][j] + bv;
            }
        }
    }
}

// ---------------- launcher ----------------
extern "C" void kernel_launch(void* const* d_in, const int* in_sizes, int n_in,
                              void* d_out, int out_size, void* d_ws, size_t ws_size,
                              hipStream_t stream) {
    const float* x      = (const float*)d_in[0];
    const float* y      = (const float*)d_in[1];
    const float* weight = (const float*)d_in[2];
    const float* bias   = (const float*)d_in[3];
    const float* fc1w   = (const float*)d_in[4];
    const float* fc1b   = (const float*)d_in[5];
    const float* fc2w   = (const float*)d_in[6];
    const float* fc2b   = (const float*)d_in[7];
    const float* p1     = (const float*)d_in[8];
    const float* p2     = (const float*)d_in[9];
    float* out = (float*)d_out;

    // workspace layout: h [16*256 f32] | wmod [16*9*128*128 bf16] (~4.75 MB total)
    float* hbuf = (float*)d_ws;
    unsigned short* wmod = (unsigned short*)((char*)d_ws + (size_t)NB * HID * sizeof(float));

    mlp1_kernel<<<NB, 256, 0, stream>>>(y, fc1w, fc1b, p1, hbuf);
    mlp2_kernel<<<(NB * COUT * CIN) / 256, 256, 0, stream>>>(hbuf, fc2w, fc2b, p2, weight, wmod);
    conv_wmma_kernel<<<NB * HH, 256, 0, stream>>>(x, wmod, bias, out);
}
